// LogMMExpDenseSPModel_22806276341805
// MI455X (gfx1250) — matrix operand order repaired and checked by reading, not compiled, not run on hardware
//
#include <hip/hip_runtime.h>
#include <hip/hip_bf16.h>
#include <cstdint>
#include <cstddef>

typedef __attribute__((ext_vector_type(16))) _Float16 v16h;
typedef __attribute__((ext_vector_type(8)))  float    v8f;

// Problem constants (match reference).
constexpr int Nn  = 2048;
constexpr int Dd  = 4096;
constexpr int Ee  = 16384;
constexpr int NUM_TILES = Ee / 16;            // 1024 column tiles of width 16
constexpr int ENT_CAP_PAD = 32 * NUM_TILES;   // max extra words from per-tile padding

// ---------------------------------------------------------------- utilities
__global__ void k_zero_i32(int* __restrict__ p, int n) {
  int i = blockIdx.x * blockDim.x + threadIdx.x;
  if (i < n) p[i] = 0;
}

// ex[i] = (f16) exp(x[i])  — precompute so the hot loop gathers f16 directly.
__global__ void k_exp_x(const float* __restrict__ x, _Float16* __restrict__ ex, int n) {
  int i = blockIdx.x * blockDim.x + threadIdx.x;
  if (i < n) ex[i] = (_Float16)__expf(x[i]);
}

// Histogram of entries per 16-wide column tile.
__global__ void k_hist(const int* __restrict__ col, int* __restrict__ counts, int nnz) {
  int i = blockIdx.x * blockDim.x + threadIdx.x;
  if (i < nnz) atomicAdd(&counts[col[i] >> 4], 1);
}

// Single-block scan over 1024 tile counts, padded to multiples of 32 so the
// main kernel's K-loop is branchless. offsets[] are padded CSR offsets;
// cursors[] are scatter cursors (start of each tile's real entries).
__global__ void k_scan(const int* __restrict__ counts, int* __restrict__ offsets,
                       int* __restrict__ cursors) {
  __shared__ int s[NUM_TILES];
  int t = threadIdx.x;
  int v  = counts[t];
  int pv = (v + 31) & ~31;          // padded count (0 stays 0 -> empty tile)
  s[t] = pv;
  __syncthreads();
  for (int d = 1; d < NUM_TILES; d <<= 1) {
    int add = (t >= d) ? s[t - d] : 0;
    __syncthreads();
    s[t] += add;
    __syncthreads();
  }
  offsets[t + 1] = s[t];
  if (t == 0) offsets[0] = 0;
  cursors[t] = s[t] - pv;           // padded exclusive prefix
}

// Scatter entries into tile-grouped (padded) order, packed into one u32:
//   bits [11:0]  row index (D=4096)
//   bits [15:12] e_local (column within 16-wide tile)
//   bits [31:16] f16 bits of exp(val)
// Padding slots keep their zero init: exp(val)=0 => B column is 0 => no effect.
__global__ void k_scatter(const float* __restrict__ vals, const int* __restrict__ rows,
                          const int* __restrict__ cols, int* __restrict__ cursors,
                          uint32_t* __restrict__ entries, int nnz) {
  int i = blockIdx.x * blockDim.x + threadIdx.x;
  if (i >= nnz) return;
  int c    = cols[i];
  int tile = c >> 4;
  int el   = c & 15;
  int pos  = atomicAdd(&cursors[tile], 1);
  union { _Float16 f; unsigned short u; } cv;
  cv.f = (_Float16)__expf(vals[i]);
  entries[pos] = (uint32_t)(rows[i] & 0xFFF) | ((uint32_t)el << 12) | ((uint32_t)cv.u << 16);
}

// ---------------------------------------------------------------- main kernel
// Block: 256 threads = 8 waves. Each block owns one 16-wide e-tile and 128 rows
// (wave w -> rows nBase + 16w .. +15). Entry list staged through LDS in chunks
// of 256 (always a multiple of 32); each wave consumes it in K=32 slices via
// v_wmma_f32_16x16x32_f16 with a fully branchless fragment build.
//
// A-fragment (16x32 f16, ISA layout): lane L (M = L%16), VGPR v half h holds
//   K = (v<4 ? 2v+h : 16+2(v-4)+h) + (L>=16 ? 8 : 0)
//   => per lane: two contiguous 8-word runs [kb+8H, +8) and [kb+16+8H, +8)
// B-fragment (32x16 f16): lane L (N = L%16), VGPR v half h holds
//   K as for A  => the same two 8-word runs [kb+8H, +8) and [kb+16+8H, +8)
// C/D (16x16 f32): VGPR r, lanes 0-15 -> M=r, lanes 16-31 -> M=r+8, N = L%16.
__global__ __launch_bounds__(256)
void k_lse_wmma(const _Float16* __restrict__ ex, const uint32_t* __restrict__ entries,
                const int* __restrict__ offsets, float* __restrict__ out) {
  const int eTile  = blockIdx.x;
  const int nBase  = blockIdx.y * 128;
  const int wave   = threadIdx.x >> 5;
  const int lane   = threadIdx.x & 31;
  const int laneM  = lane & 15;
  const int hiHalf = (lane >> 4) & 1;

  const int beg = offsets[eTile];
  const int end = offsets[eTile + 1];

  __shared__ uint32_t sEnt[256];

  const uint32_t rowOff = (uint32_t)(nBase + wave * 16 + laneM) * (uint32_t)Dd;

  v8f acc = {};

  for (int base = beg; base < end; base += 256) {
    const int cnt = min(end - base, 256);   // always a multiple of 32
    // stage this chunk of the tile's entry list into LDS (shared by 8 waves)
    for (int i = (int)threadIdx.x; i < cnt; i += 256) sEnt[i] = entries[base + i];
    // prefetch the next chunk while we compute (global_prefetch_b8)
    if (base + 256 < end) __builtin_prefetch(entries + base + 256, 0, 1);
    __syncthreads();

    for (int kb = 0; kb < cnt; kb += 32) {
      // --- vector LDS reads of the entry words this lane needs (ds_load_b128)
      const uint32_t* pA = &sEnt[kb + 8 * hiHalf];
      const uint4 ra0 = *(const uint4*)(pA);
      const uint4 ra1 = *(const uint4*)(pA + 4);
      const uint4 ra2 = *(const uint4*)(pA + 16);
      const uint4 ra3 = *(const uint4*)(pA + 20);
      const uint32_t* pB = &sEnt[kb + 8 * hiHalf];
      const uint4 rb0 = *(const uint4*)(pB);
      const uint4 rb1 = *(const uint4*)(pB + 4);
      const uint4 rb2 = *(const uint4*)(pB + 16);
      const uint4 rb3 = *(const uint4*)(pB + 20);

      const uint32_t eA[16] = {ra0.x, ra0.y, ra0.z, ra0.w, ra1.x, ra1.y, ra1.z, ra1.w,
                               ra2.x, ra2.y, ra2.z, ra2.w, ra3.x, ra3.y, ra3.z, ra3.w};
      const uint32_t eB[16] = {rb0.x, rb0.y, rb0.z, rb0.w, rb1.x, rb1.y, rb1.z, rb1.w,
                               rb2.x, rb2.y, rb2.z, rb2.w, rb3.x, rb3.y, rb3.z, rb3.w};

      // --- batch the 16 independent gathers of exp(x[n, row_k]) (uniform base
      //     + 32-bit voffset so they cluster under one loadcnt wait)
      uint32_t offA[16];
#pragma unroll
      for (int p = 0; p < 16; ++p) offA[p] = rowOff + (eA[p] & 0xFFFu);
      _Float16 av[16];
#pragma unroll
      for (int p = 0; p < 16; ++p) av[p] = ex[offA[p]];

      // --- assemble fragments (pure VALU; padding entries give B == 0)
      v16h a = {};
      v16h b = {};
#pragma unroll
      for (int p = 0; p < 16; ++p) {
        a[p] = av[p];
        const uint32_t m = eB[p];
        union { unsigned short u; _Float16 f; } cv;
        cv.u = (unsigned short)(m >> 16);
        b[p] = (((m >> 12) & 0xFu) == (uint32_t)laneM) ? cv.f : (_Float16)0.f;
      }

      // segmented sum-of-exp accumulates in the f32 accumulator
      acc = __builtin_amdgcn_wmma_f32_16x16x32_f16(
          false, a, false, b, (short)0, acc, false, false);
    }
    __syncthreads();
  }

  // log(sums); empty column -> log(0) = -inf, matching the reference.
  const int colE = eTile * 16 + laneM;
#pragma unroll
  for (int r = 0; r < 8; ++r) {
    const int m = r + (hiHalf ? 8 : 0);
    const int n = nBase + wave * 16 + m;
    out[(size_t)n * Ee + colE] = logf(acc[r]);
  }
}

// ---------------------------------------------------------------- launcher
extern "C" void kernel_launch(void* const* d_in, const int* in_sizes, int n_in,
                              void* d_out, int out_size, void* d_ws, size_t ws_size,
                              hipStream_t stream) {
  const float* x     = (const float*)d_in[0];
  const float* avals = (const float*)d_in[1];
  const int*   arow  = (const int*)d_in[2];
  const int*   acol  = (const int*)d_in[3];
  const int    nnz   = in_sizes[1];
  float*       out   = (float*)d_out;

  const int entCap = nnz + ENT_CAP_PAD;   // capacity incl. per-tile padding

  // Workspace layout (~16.6 MiB total)
  char* ws = (char*)d_ws;
  size_t off = 0;
  _Float16* ex = (_Float16*)(ws + off);           off += (size_t)Nn * Dd * sizeof(_Float16);
  int* counts  = (int*)(ws + off);                off += (size_t)NUM_TILES * sizeof(int);
  int* offsets = (int*)(ws + off);                off += (size_t)(NUM_TILES + 1) * sizeof(int);
  int* cursors = (int*)(ws + off);                off += (size_t)NUM_TILES * sizeof(int);
  off = (off + 255) & ~(size_t)255;
  uint32_t* entries = (uint32_t*)(ws + off);      off += (size_t)entCap * sizeof(uint32_t);

  // 1) zero tile histogram and the (padded) entries buffer
  k_zero_i32<<<(NUM_TILES + 255) / 256, 256, 0, stream>>>(counts, NUM_TILES);
  k_zero_i32<<<(entCap + 255) / 256, 256, 0, stream>>>((int*)entries, entCap);
  // 2) ex = (f16) exp(x)
  {
    const int total = Nn * Dd;
    k_exp_x<<<(total + 255) / 256, 256, 0, stream>>>(x, ex, total);
  }
  // 3) histogram of entries per 16-wide column tile
  k_hist<<<(nnz + 255) / 256, 256, 0, stream>>>(acol, counts, nnz);
  // 4) scan -> padded CSR offsets + scatter cursors
  k_scan<<<1, NUM_TILES, 0, stream>>>(counts, offsets, cursors);
  // 5) scatter packed entries into tile-grouped (padded) order
  k_scatter<<<(nnz + 255) / 256, 256, 0, stream>>>(avals, arow, acol, cursors, entries, nnz);
  // 6) main WMMA segmented-logsumexp kernel
  {
    dim3 grid(NUM_TILES, Nn / 128);   // 1024 x 16 blocks
    k_lse_wmma<<<grid, 256, 0, stream>>>(ex, entries, offsets, out);
  }
}
